// HierarchicalMessagePassing_89936615178810
// MI455X (gfx1250) — matrix operand order repaired and checked by reading, not compiled, not run on hardware
//
#include <hip/hip_runtime.h>
#include <hip/hip_bf16.h>
#include <math.h>

typedef __attribute__((ext_vector_type(16))) __bf16 v16bf;
typedef __attribute__((ext_vector_type(8)))  __bf16 v8bf;
typedef __attribute__((ext_vector_type(8)))  float  v8f;

#define NCELLS 131072
#define NREG   512
#define DDIM   512
#define HDIM   256

__device__ __forceinline__ float gelu_exact(float x) {
    return 0.5f * x * (1.0f + erff(x * 0.70710678118654752f));
}
__device__ __forceinline__ float sigmoid_(float x) { return 1.0f / (1.0f + expf(-x)); }

// ---------------------------------------------------------------------------
// Prep: bf16 conversions, weight transposes (B-layout: row = out col, contig K),
// zero accumulators. Grid covers R*D = 262144 elements.
// ---------------------------------------------------------------------------
__global__ void k_prep(const float* __restrict__ protos,
                       const float* __restrict__ w_ua, const float* __restrict__ w_up,
                       const float* __restrict__ w_d1, const float* __restrict__ w_d2,
                       __bf16* __restrict__ protos_bf, __bf16* __restrict__ wua_t_bf,
                       float* __restrict__ wup_t, float* __restrict__ wd1_t,
                       float* __restrict__ wd2_t,
                       float* __restrict__ agg, unsigned* __restrict__ seg_max,
                       float* __restrict__ seg_sum, unsigned* __restrict__ counts)
{
    int idx = blockIdx.x * 256 + threadIdx.x;
    if (idx < NREG * DDIM) protos_bf[idx] = (__bf16)protos[idx];
    if (idx < DDIM * HDIM) {
        int k = idx / HDIM, h = idx % HDIM;            // k also serves as d (0..511)
        wua_t_bf[h * DDIM + k] = (__bf16)w_ua[k * HDIM + h];  // w_ua [D,H] -> [H,D]
        wd1_t[h * DDIM + k]    = w_d1[k * HDIM + h];          // w_d1 [D,H] -> [H,D]
        wup_t[k * HDIM + h]    = w_up[h * DDIM + k];          // w_up [H,D] -> [D,H]
        wd2_t[k * HDIM + h]    = w_d2[h * DDIM + k];          // w_d2 [H,D] -> [D,H]
    }
    if (idx < NREG * HDIM) agg[idx] = 0.0f;
    if (idx < NREG) { seg_max[idx] = 0u; seg_sum[idx] = 0.0f; counts[idx] = 0u; }
}

// ---------------------------------------------------------------------------
// Row-softmax GEMM: 16 cell rows per block, logits vs 512 columns via WMMA bf16.
// Writes softmax probability at each row's own region. mode==0: also segment
// max (bit-pattern atomicMax, w>0) and counts for the up pass.
// ---------------------------------------------------------------------------
__global__ __launch_bounds__(128) void k_rowsoftmax(
    const float* __restrict__ zl, const __bf16* __restrict__ Bm,
    const int* __restrict__ regions, const float* __restrict__ temp_p,
    float* __restrict__ w_out, unsigned* __restrict__ seg_max_bits,
    unsigned* __restrict__ counts, int mode)
{
    __shared__ __align__(16) __bf16 ldsA[16][DDIM + 8];
    __shared__ float ldsL[16][NREG + 1];
    __shared__ float red[16][8];
    __shared__ float rowm[16], rowsum[16];

    const int tid  = threadIdx.x;
    const int wave = tid >> 5, lane = tid & 31;
    const int half = lane >> 4, rlane = lane & 15;
    const long rowBase = (long)blockIdx.x * 16;

    // Stage A tile (16 x 512 f32 -> bf16) into LDS.
    for (int it = tid; it < 16 * 128; it += 128) {
        int row = it >> 7, c4 = (it & 127) << 2;
        const float4 v = *(const float4*)(zl + (rowBase + row) * DDIM + c4);
        ldsA[row][c4 + 0] = (__bf16)v.x; ldsA[row][c4 + 1] = (__bf16)v.y;
        ldsA[row][c4 + 2] = (__bf16)v.z; ldsA[row][c4 + 3] = (__bf16)v.w;
    }
    __syncthreads();
    const float invtemp = 1.0f / temp_p[0];

    // 32 column tiles of 16; 4 waves x 8 tiles.
    for (int j = 0; j < 8; ++j) {
        const int ct = wave * 8 + j;
        v8f acc = {0.f, 0.f, 0.f, 0.f, 0.f, 0.f, 0.f, 0.f};
        #pragma unroll
        for (int kt = 0; kt < 16; ++kt) {
            const int k0 = kt * 32 + half * 8;        // A elems 0..7
            const int k1 = kt * 32 + 16 + half * 8;   // A elems 8..15
            v8bf alo = *(const v8bf*)&ldsA[rlane][k0];
            v8bf ahi = *(const v8bf*)&ldsA[rlane][k1];
            v16bf a;
            #pragma unroll
            for (int e = 0; e < 8; ++e) { a[e] = alo[e]; a[e + 8] = ahi[e]; }
            // B: lane n = rlane, K = kt*32 + half*8 .. +7 and kt*32 + 16 + half*8 .. +7
            const __bf16* bp = Bm + (size_t)(ct * 16 + rlane) * DDIM + kt * 32 + half * 8;
            v8bf blo = *(const v8bf*)bp;
            v8bf bhi = *(const v8bf*)(bp + 16);
            v16bf b;
            #pragma unroll
            for (int e = 0; e < 8; ++e) { b[e] = blo[e]; b[e + 8] = bhi[e]; }
            acc = __builtin_amdgcn_wmma_f32_16x16x32_bf16(false, a, false, b,
                                                          (short)0, acc, false, false);
        }
        #pragma unroll
        for (int g = 0; g < 8; ++g)
            ldsL[half * 8 + g][ct * 16 + rlane] = acc[g] * invtemp;
    }
    __syncthreads();

    // Row softmax: 8 threads/row, 64 columns each.
    const int row = tid >> 3, seg = tid & 7;
    float mx = -3.4e38f;
    for (int c = 0; c < 64; ++c) mx = fmaxf(mx, ldsL[row][seg * 64 + c]);
    red[row][seg] = mx;
    __syncthreads();
    if (seg == 0) {
        float m = red[row][0];
        for (int s = 1; s < 8; ++s) m = fmaxf(m, red[row][s]);
        rowm[row] = m;
    }
    __syncthreads();
    const float m = rowm[row];
    float s = 0.f;
    for (int c = 0; c < 64; ++c) s += expf(ldsL[row][seg * 64 + c] - m);
    red[row][seg] = s;
    __syncthreads();
    if (seg == 0) {
        float t = 0.f;
        for (int ss = 0; ss < 8; ++ss) t += red[row][ss];
        rowsum[row] = t;
    }
    __syncthreads();
    if (tid < 16) {
        const long i = rowBase + tid;
        const int r = regions[i];
        const float pw = expf(ldsL[tid][r] - rowm[tid]) / rowsum[tid];
        w_out[i] = pw;
        if (mode == 0) {
            atomicMax(&seg_max_bits[r], __float_as_uint(pw));
            atomicAdd(&counts[r], 1u);
        }
    }
}

// ---------------------------------------------------------------------------
// Segment softmax: e_i = exp(w_i - m[r_i]); seg_sum += e_i
// ---------------------------------------------------------------------------
__global__ void k_segexp(const float* __restrict__ w, const int* __restrict__ regions,
                         const unsigned* __restrict__ seg_max_bits,
                         float* __restrict__ e_out, float* __restrict__ seg_sum)
{
    const long i = (long)blockIdx.x * 256 + threadIdx.x;
    const int r = regions[i];
    const float mv = __uint_as_float(seg_max_bits[r]);
    const float e = expf(w[i] - mv);
    e_out[i] = e;
    atomicAdd(&seg_sum[r], e);
}

// ---------------------------------------------------------------------------
// Up aggregation: h = LN(GELU(z_local @ w_ua + b_ua)); agg += h * w_norm.
// WMMA bf16 GEMM (16 rows x 256 cols per block) + fused epilogue + atomics.
// ---------------------------------------------------------------------------
__global__ __launch_bounds__(128) void k_up_agg(
    const float* __restrict__ zl, const __bf16* __restrict__ Bm, // wua_t [H,D] bf16
    const float* __restrict__ b_ua, const float* __restrict__ g_ua,
    const float* __restrict__ be_ua, const int* __restrict__ regions,
    const float* __restrict__ e_buf, const float* __restrict__ seg_sum,
    float* __restrict__ agg)
{
    __shared__ __align__(16) __bf16 ldsA[16][DDIM + 8];
    __shared__ float ldsH[16][HDIM + 1];
    __shared__ float redA[16][8], redB[16][8];
    __shared__ float rmu[16], rrs[16];

    const int tid  = threadIdx.x;
    const int wave = tid >> 5, lane = tid & 31;
    const int half = lane >> 4, rlane = lane & 15;
    const long rowBase = (long)blockIdx.x * 16;

    for (int it = tid; it < 16 * 128; it += 128) {
        int row = it >> 7, c4 = (it & 127) << 2;
        const float4 v = *(const float4*)(zl + (rowBase + row) * DDIM + c4);
        ldsA[row][c4 + 0] = (__bf16)v.x; ldsA[row][c4 + 1] = (__bf16)v.y;
        ldsA[row][c4 + 2] = (__bf16)v.z; ldsA[row][c4 + 3] = (__bf16)v.w;
    }
    __syncthreads();

    for (int j = 0; j < 4; ++j) {                      // 16 tiles: 4 waves x 4
        const int ct = wave * 4 + j;
        v8f acc = {0.f, 0.f, 0.f, 0.f, 0.f, 0.f, 0.f, 0.f};
        #pragma unroll
        for (int kt = 0; kt < 16; ++kt) {
            const int k0 = kt * 32 + half * 8;
            const int k1 = kt * 32 + 16 + half * 8;
            v8bf alo = *(const v8bf*)&ldsA[rlane][k0];
            v8bf ahi = *(const v8bf*)&ldsA[rlane][k1];
            v16bf a;
            #pragma unroll
            for (int e = 0; e < 8; ++e) { a[e] = alo[e]; a[e + 8] = ahi[e]; }
            const __bf16* bp = Bm + (size_t)(ct * 16 + rlane) * DDIM + kt * 32 + half * 8;
            v8bf blo = *(const v8bf*)bp;
            v8bf bhi = *(const v8bf*)(bp + 16);
            v16bf b;
            #pragma unroll
            for (int e = 0; e < 8; ++e) { b[e] = blo[e]; b[e + 8] = bhi[e]; }
            acc = __builtin_amdgcn_wmma_f32_16x16x32_bf16(false, a, false, b,
                                                          (short)0, acc, false, false);
        }
        #pragma unroll
        for (int g = 0; g < 8; ++g) {
            const int col = ct * 16 + rlane;
            ldsH[half * 8 + g][col] = gelu_exact(acc[g] + b_ua[col]);
        }
    }
    __syncthreads();

    // LayerNorm over H=256 per row: 8 threads/row, 32 cols each.
    const int row = tid >> 3, seg = tid & 7;
    float s = 0.f, s2 = 0.f;
    for (int c = 0; c < 32; ++c) {
        const float x = ldsH[row][seg * 32 + c];
        s += x; s2 += x * x;
    }
    redA[row][seg] = s; redB[row][seg] = s2;
    __syncthreads();
    if (seg == 0) {
        float S = 0.f, S2 = 0.f;
        for (int ss = 0; ss < 8; ++ss) { S += redA[row][ss]; S2 += redB[row][ss]; }
        const float mu = S / (float)HDIM;
        const float var = S2 / (float)HDIM - mu * mu;
        rmu[row] = mu; rrs[row] = rsqrtf(var + 1e-5f);
    }
    __syncthreads();
    const long i = rowBase + row;
    const int r = regions[i];
    const float wn = e_buf[i] / seg_sum[r];
    const float mu = rmu[row], rs = rrs[row];
    for (int c = 0; c < 32; ++c) {
        const int col = seg * 32 + c;
        const float x = ldsH[row][col];
        const float v = ((x - mu) * rs * g_ua[col] + be_ua[col]) * wn;
        atomicAdd(&agg[(size_t)r * HDIM + col], v);
    }
}

// ---------------------------------------------------------------------------
// Region update: upd = LN(GELU(agg @ w_up + b_up)); updated_z_fused.
// ---------------------------------------------------------------------------
__global__ __launch_bounds__(256) void k_region_update(
    const float* __restrict__ agg, const float* __restrict__ wup_t,
    const float* __restrict__ b_up, const float* __restrict__ g_up,
    const float* __restrict__ be_up, const unsigned* __restrict__ counts,
    const float* __restrict__ z_fused,
    float* __restrict__ updated_zf, __bf16* __restrict__ zf_bf)
{
    __shared__ float aggRow[HDIM];
    __shared__ float xbuf[DDIM];
    __shared__ float ps[256], ps2[256];
    __shared__ float mu_s, rs_s;
    const int r = blockIdx.x, tid = threadIdx.x;

    aggRow[tid] = agg[(size_t)r * HDIM + tid];
    __syncthreads();
    #pragma unroll
    for (int p = 0; p < 2; ++p) {
        const int t = tid + p * 256;
        float acc = b_up[t];
        for (int h = 0; h < HDIM; ++h) acc += aggRow[h] * wup_t[(size_t)t * HDIM + h];
        xbuf[t] = gelu_exact(acc);
    }
    __syncthreads();
    {
        const float a = xbuf[tid], b = xbuf[tid + 256];
        ps[tid] = a + b; ps2[tid] = a * a + b * b;
    }
    __syncthreads();
    for (int st = 128; st > 0; st >>= 1) {
        if (tid < st) { ps[tid] += ps[tid + st]; ps2[tid] += ps2[tid + st]; }
        __syncthreads();
    }
    if (tid == 0) {
        const float mu = ps[0] / (float)DDIM;
        const float var = ps2[0] / (float)DDIM - mu * mu;
        mu_s = mu; rs_s = rsqrtf(var + 1e-5f);
    }
    __syncthreads();
    const unsigned cnt = counts[r];
    #pragma unroll
    for (int p = 0; p < 2; ++p) {
        const int t = tid + p * 256;
        const float upd = (xbuf[t] - mu_s) * rs_s * g_up[t] + be_up[t];
        const float out = (cnt > 0u) ? upd : z_fused[(size_t)r * DDIM + t];
        updated_zf[(size_t)r * DDIM + t] = out;
        zf_bf[(size_t)r * DDIM + t] = (__bf16)out;
    }
}

// ---------------------------------------------------------------------------
// Down broadcast (region MLP): hb = LN(GELU(zf @ w_d1)); bcast = hb @ w_d2 + b_d2
// ---------------------------------------------------------------------------
__global__ __launch_bounds__(512) void k_down_broadcast(
    const float* __restrict__ updated_zf,
    const float* __restrict__ wd1_t, const float* __restrict__ b_d1,
    const float* __restrict__ g_d1, const float* __restrict__ be_d1,
    const float* __restrict__ wd2_t, const float* __restrict__ b_d2,
    float* __restrict__ bcast)
{
    __shared__ float zfRow[DDIM];
    __shared__ float xs[HDIM];
    __shared__ float hbRow[HDIM];
    __shared__ float ps[256], ps2[256];
    __shared__ float mu_s, rs_s;
    const int r = blockIdx.x, tid = threadIdx.x;

    zfRow[tid] = updated_zf[(size_t)r * DDIM + tid];
    __syncthreads();
    if (tid < HDIM) {
        float acc = b_d1[tid];
        for (int k = 0; k < DDIM; ++k) acc += zfRow[k] * wd1_t[(size_t)tid * DDIM + k];
        xs[tid] = gelu_exact(acc);
    }
    __syncthreads();
    if (tid < 256) { const float x = xs[tid]; ps[tid] = x; ps2[tid] = x * x; }
    __syncthreads();
    for (int st = 128; st > 0; st >>= 1) {
        if (tid < st) { ps[tid] += ps[tid + st]; ps2[tid] += ps2[tid + st]; }
        __syncthreads();
    }
    if (tid == 0) {
        const float mu = ps[0] / (float)HDIM;
        const float var = ps2[0] / (float)HDIM - mu * mu;
        mu_s = mu; rs_s = rsqrtf(var + 1e-5f);
    }
    __syncthreads();
    if (tid < HDIM) hbRow[tid] = (xs[tid] - mu_s) * rs_s * g_d1[tid] + be_d1[tid];
    __syncthreads();
    float acc = b_d2[tid];
    for (int h = 0; h < HDIM; ++h) acc += hbRow[h] * wd2_t[(size_t)tid * HDIM + h];
    bcast[(size_t)r * DDIM + tid] = acc;
}

// ---------------------------------------------------------------------------
// Final blends
// ---------------------------------------------------------------------------
__global__ void k_final_local(const float* __restrict__ zl, const int* __restrict__ regions,
                              const float* __restrict__ weight, const float* __restrict__ bcast,
                              const float* __restrict__ raw_rw_p, float* __restrict__ out)
{
    const long idx4 = (long)blockIdx.x * 256 + threadIdx.x;   // over N*D/4
    const long i = idx4 >> 7;
    const int d = (int)(idx4 & 127) << 2;
    const float rw = sigmoid_(raw_rw_p[0]);
    const int r = regions[i];
    const float wg = weight[i];
    const float4 z = *(const float4*)(zl + i * DDIM + d);
    const float4 b = *(const float4*)(bcast + (size_t)r * DDIM + d);
    float4 o;
    o.x = rw * (b.x * wg) + (1.f - rw) * z.x;
    o.y = rw * (b.y * wg) + (1.f - rw) * z.y;
    o.z = rw * (b.z * wg) + (1.f - rw) * z.z;
    o.w = rw * (b.w * wg) + (1.f - rw) * z.w;
    *(float4*)(out + i * DDIM + d) = o;
}

__global__ void k_final_fused(const float* __restrict__ z_fused,
                              const float* __restrict__ updated_zf,
                              const float* __restrict__ raw_rw_p, float* __restrict__ out)
{
    const int idx = blockIdx.x * 256 + threadIdx.x;           // over R*D
    const float rw = sigmoid_(raw_rw_p[0]);
    out[idx] = rw * updated_zf[idx] + (1.f - rw) * z_fused[idx];
}

// ---------------------------------------------------------------------------
extern "C" void kernel_launch(void* const* d_in, const int* in_sizes, int n_in,
                              void* d_out, int out_size, void* d_ws, size_t ws_size,
                              hipStream_t stream) {
    const float* z_local = (const float*)d_in[0];
    const float* z_fused = (const float*)d_in[1];
    const int*   regions = (const int*)d_in[2];
    const float* protos  = (const float*)d_in[3];
    const float* temp_p  = (const float*)d_in[4];
    const float* rawrw_p = (const float*)d_in[5];
    const float* w_ua = (const float*)d_in[6];
    const float* b_ua = (const float*)d_in[7];
    const float* g_ua = (const float*)d_in[8];
    const float* be_ua = (const float*)d_in[9];
    const float* w_up = (const float*)d_in[10];
    const float* b_up = (const float*)d_in[11];
    const float* g_up = (const float*)d_in[12];
    const float* be_up = (const float*)d_in[13];
    const float* w_d1 = (const float*)d_in[14];
    const float* b_d1 = (const float*)d_in[15];
    const float* g_d1 = (const float*)d_in[16];
    const float* be_d1 = (const float*)d_in[17];
    const float* w_d2 = (const float*)d_in[18];
    const float* b_d2 = (const float*)d_in[19];

    char* p = (char*)d_ws;
    auto alloc = [&](size_t bytes) -> void* {
        void* q = (void*)p;
        p += (bytes + 255) & ~(size_t)255;
        return q;
    };
    __bf16*   protos_bf = (__bf16*)  alloc((size_t)NREG * DDIM * 2);
    __bf16*   wua_t_bf  = (__bf16*)  alloc((size_t)HDIM * DDIM * 2);
    float*    wup_t     = (float*)   alloc((size_t)DDIM * HDIM * 4);
    float*    wd1_t     = (float*)   alloc((size_t)HDIM * DDIM * 4);
    float*    wd2_t     = (float*)   alloc((size_t)DDIM * HDIM * 4);
    float*    w_buf     = (float*)   alloc((size_t)NCELLS * 4);
    float*    e_buf     = (float*)   alloc((size_t)NCELLS * 4);
    float*    weight    = (float*)   alloc((size_t)NCELLS * 4);
    unsigned* seg_max   = (unsigned*)alloc((size_t)NREG * 4);
    float*    seg_sum   = (float*)   alloc((size_t)NREG * 4);
    unsigned* counts    = (unsigned*)alloc((size_t)NREG * 4);
    float*    agg       = (float*)   alloc((size_t)NREG * HDIM * 4);
    float*    upd_zf    = (float*)   alloc((size_t)NREG * DDIM * 4);
    __bf16*   zf_bf     = (__bf16*)  alloc((size_t)NREG * DDIM * 2);
    float*    bcast     = (float*)   alloc((size_t)NREG * DDIM * 4);

    float* out_local = (float*)d_out;                 // [N,D]
    float* out_fused = (float*)d_out + (size_t)NCELLS * DDIM;  // [R,D]

    // 1) prep + zero accumulators
    k_prep<<<(NREG * DDIM) / 256, 256, 0, stream>>>(
        protos, w_ua, w_up, w_d1, w_d2,
        protos_bf, wua_t_bf, wup_t, wd1_t, wd2_t,
        agg, seg_max, seg_sum, counts);

    // 2) up logits + row softmax -> own-region prob w; segment max + counts
    k_rowsoftmax<<<NCELLS / 16, 128, 0, stream>>>(
        z_local, protos_bf, regions, temp_p, w_buf, seg_max, counts, 0);

    // 3) segment exp/sum
    k_segexp<<<NCELLS / 256, 256, 0, stream>>>(w_buf, regions, seg_max, e_buf, seg_sum);

    // 4) up aggregation GEMM + fused GELU/LN + weighted segment scatter
    k_up_agg<<<NCELLS / 16, 128, 0, stream>>>(
        z_local, wua_t_bf, b_ua, g_ua, be_ua, regions, e_buf, seg_sum, agg);

    // 5) region update MLP
    k_region_update<<<NREG, 256, 0, stream>>>(
        agg, wup_t, b_up, g_up, be_up, counts, z_fused, upd_zf, zf_bf);

    // 6) down logits + row softmax -> receive weight
    k_rowsoftmax<<<NCELLS / 16, 128, 0, stream>>>(
        z_local, zf_bf, regions, temp_p, weight, seg_max, counts, 1);

    // 7) down broadcast MLP
    k_down_broadcast<<<NREG, 512, 0, stream>>>(
        upd_zf, wd1_t, b_d1, g_d1, be_d1, wd2_t, b_d2, bcast);

    // 8) final blends
    k_final_local<<<((size_t)NCELLS * DDIM / 4) / 256, 256, 0, stream>>>(
        z_local, regions, weight, bcast, rawrw_p, out_local);
    k_final_fused<<<(NREG * DDIM) / 256, 256, 0, stream>>>(
        z_fused, upd_zf, rawrw_p, out_fused);
}
